// VDTBARTransformer_41180146434495
// MI455X (gfx1250) — compile-verified, matrix operand order checked by reading, not run on hardware
//
#include <hip/hip_runtime.h>
#include <hip/hip_bf16.h>

// ---------------------------------------------------------------------------
// GPT-2-style transformer forward (L=4, D=1024, H=16, HD=64, DF=4096, B=2,
// T=2048, fp32 in/out) for gfx1250 (MI455X).
//
// Compute-bound workload (~550 GFLOP vs ~8us of HBM traffic at 23.3 TB/s):
// every matmul runs on v_wmma_f32_16x16x32_bf16 with bf16 operands end-to-end
// (weights converted once per launch; LayerNorm / QKV / fc / attention all
// emit bf16 directly), fp32 accumulation, fp32 residual stream.
// Attention exploits causality (only key tiles kt <= qt are processed, and
// only the diagonal tile pays for masking).  The GEMM k-loop is software-
// pipelined: the next global tile is staged into registers while the current
// tile's 8 WMMAs execute.
// ---------------------------------------------------------------------------

typedef __attribute__((ext_vector_type(16))) __bf16          v16bf;
typedef __attribute__((ext_vector_type(16))) unsigned short  v16u;
typedef __attribute__((ext_vector_type(8)))  float           v8f;

#define Lc  4
#define Dc  1024
#define Hc  16
#define HDc 64
#define DFc 4096
#define Bc  2
#define Tc  2048
#define MTc (Bc * Tc)          // 4096 tokens

__device__ __forceinline__ unsigned short f2bf(float f) {
  unsigned int u = __builtin_bit_cast(unsigned int, f);
  unsigned int r = u + 0x7FFFu + ((u >> 16) & 1u);   // round-to-nearest-even
  return (unsigned short)(r >> 16);
}

__device__ __forceinline__ v8f wmma_bf16(v16u a, v16u b, v8f c) {
  return __builtin_amdgcn_wmma_f32_16x16x32_bf16(
      /*neg_a=*/false, __builtin_bit_cast(v16bf, a),
      /*neg_b=*/false, __builtin_bit_cast(v16bf, b),
      /*c_mod=*/(short)0, c, /*reuse_a=*/false, /*reuse_b=*/false);
}

// ---------------------------------------------------------------------------
// fp32 -> bf16 bulk conversion (weights; once per launch, memory-bound).
// ---------------------------------------------------------------------------
__global__ __launch_bounds__(256) void cvt_kernel(
    const float* __restrict__ in, unsigned short* __restrict__ out, long n4) {
  for (long i = (long)blockIdx.x * 256 + threadIdx.x; i < n4;
       i += (long)gridDim.x * 256) {
    float4 v = ((const float4*)in)[i];
    ushort4 o;
    o.x = f2bf(v.x); o.y = f2bf(v.y); o.z = f2bf(v.z); o.w = f2bf(v.w);
    ((ushort4*)out)[i] = o;
  }
}

// ---------------------------------------------------------------------------
// LayerNorm: one 256-thread block per token row of D floats.
// Writes bf16 (yb) when feeding a GEMM, or fp32 (yf) for the final output.
// ---------------------------------------------------------------------------
__global__ __launch_bounds__(256) void ln_kernel(
    const float* __restrict__ x, const float* __restrict__ w,
    const float* __restrict__ b, float* __restrict__ yf,
    unsigned short* __restrict__ yb, int D) {
  __shared__ float red[256];
  const int t = threadIdx.x;
  const float* xr = x + (size_t)blockIdx.x * D;

  float s = 0.f;
  for (int i = t; i < D; i += 256) s += xr[i];
  red[t] = s;
  __syncthreads();
  for (int o = 128; o > 0; o >>= 1) {
    if (t < o) red[t] += red[t + o];
    __syncthreads();
  }
  const float mean = red[0] / (float)D;
  __syncthreads();

  float v = 0.f;
  for (int i = t; i < D; i += 256) {
    float d = xr[i] - mean;
    v += d * d;
  }
  red[t] = v;
  __syncthreads();
  for (int o = 128; o > 0; o >>= 1) {
    if (t < o) red[t] += red[t + o];
    __syncthreads();
  }
  const float rstd = rsqrtf(red[0] / (float)D + 1e-5f);

  for (int i = t; i < D; i += 256) {
    float o = (xr[i] - mean) * rstd * w[i] + b[i];
    if (yb) yb[(size_t)blockIdx.x * D + i] = f2bf(o);
    else    yf[(size_t)blockIdx.x * D + i] = o;
  }
}

// ---------------------------------------------------------------------------
// GEMM: Out[M,N] = epilogue(A[M,K] * W[K,N] + bias).  A, W bf16; acc fp32.
// 128x128 block tile, BK=32, 8 waves (4M x 2N), each wave 32x64 (2x4 frags).
// Software-pipelined: next k-tile staged to registers during current WMMAs.
// MODE 0: fp32 out + fp32 residual add   (proj / fc2)
// MODE 1: bf16 out                        (QKV)
// MODE 2: bf16 out + ReLU                 (fc)
// ---------------------------------------------------------------------------
#define BM 128
#define BN 128
#define BK 32

template <int MODE>
__global__ __launch_bounds__(256) void gemm_bf16_kernel(
    const unsigned short* __restrict__ A, const unsigned short* __restrict__ W,
    const float* __restrict__ bias, const float* __restrict__ Res,
    float* __restrict__ OutF, unsigned short* __restrict__ OutB,
    int M, int N, int K) {
  __shared__ __align__(32) unsigned short As[BM * BK];   // 8 KB, [m][k]
  __shared__ __align__(32) unsigned short Ws[BN * BK];   // 8 KB, [n][k]

  const int t    = threadIdx.x;
  const int lane = t & 31;
  const int w    = t >> 5;
  const int wm   = w & 3;
  const int wn   = w >> 2;
  const int bm   = blockIdx.y;
  const int bn   = blockIdx.x;
  const int hf   = lane >> 4;
  const int lr   = lane & 15;

  v8f acc[2][4];
#pragma unroll
  for (int i = 0; i < 2; i++)
#pragma unroll
    for (int j = 0; j < 4; j++) acc[i][j] = (v8f){0, 0, 0, 0, 0, 0, 0, 0};

  // register staging for the software pipeline
  uint4 rA[2], rW[2];
  auto stage = [&](int k0) {
#pragma unroll
    for (int i = 0; i < 2; i++) {
      int f = t + i * 256;                       // 16B segment 0..511
      rA[i] = *(const uint4*)(A + (size_t)(bm * BM + (f >> 2)) * K + k0 + (f & 3) * 8);
      rW[i] = *(const uint4*)(W + (size_t)(k0 + (f >> 4)) * N + bn * BN + (f & 15) * 8);
    }
  };
  stage(0);

  for (int k0 = 0; k0 < K; k0 += BK) {
    // --- commit staged tiles to LDS (A straight copy, W transposed [n][k]) --
#pragma unroll
    for (int i = 0; i < 2; i++) {
      int f = t + i * 256;
      *(uint4*)(As + (f >> 2) * BK + (f & 3) * 8) = rA[i];
      int kr  = f >> 4;
      int seg = f & 15;
      uint4 v = rW[i];
      unsigned short* wp = Ws + (seg * 8) * BK + kr;
      wp[0 * BK] = (unsigned short)(v.x);  wp[1 * BK] = (unsigned short)(v.x >> 16);
      wp[2 * BK] = (unsigned short)(v.y);  wp[3 * BK] = (unsigned short)(v.y >> 16);
      wp[4 * BK] = (unsigned short)(v.z);  wp[5 * BK] = (unsigned short)(v.z >> 16);
      wp[6 * BK] = (unsigned short)(v.w);  wp[7 * BK] = (unsigned short)(v.w >> 16);
    }
    __syncthreads();

    // --- prefetch next k-tile while WMMAs run -------------------------------
    if (k0 + BK < K) stage(k0 + BK);

    v16u a[2], bb[4];
#pragma unroll
    for (int i = 0; i < 2; i++)
      a[i] = *(const v16u*)(As + (wm * 32 + i * 16 + lr) * BK + hf * 16);
#pragma unroll
    for (int j = 0; j < 4; j++)
      bb[j] = *(const v16u*)(Ws + (wn * 64 + j * 16 + lr) * BK + hf * 16);
#pragma unroll
    for (int i = 0; i < 2; i++)
#pragma unroll
      for (int j = 0; j < 4; j++) acc[i][j] = wmma_bf16(a[i], bb[j], acc[i][j]);
    __syncthreads();
  }

  // --- epilogue (branch-free, compile-time MODE) ----------------------------
#pragma unroll
  for (int i = 0; i < 2; i++) {
#pragma unroll
    for (int j = 0; j < 4; j++) {
      const int col = bn * BN + wn * 64 + j * 16 + lr;
      const float bc = bias[col];
#pragma unroll
      for (int r = 0; r < 8; r++) {
        const int row = bm * BM + wm * 32 + i * 16 + hf * 8 + r;
        const size_t idx = (size_t)row * N + col;
        float val = acc[i][j][r] + bc;
        if (MODE == 2) val = fmaxf(val, 0.f);
        if (MODE == 0) OutF[idx] = val + Res[idx];
        else           OutB[idx] = f2bf(val);
      }
    }
  }
}

// ---------------------------------------------------------------------------
// Attention: one wave per (b, h, 16-query tile).  Two-pass flash-style over
// bf16 qkv ([B,T,3D], q|k|v at 0|D|2D), causally pruned: only key tiles
// kt <= qt are touched; only the diagonal tile pays for mask compares.
// ---------------------------------------------------------------------------
__device__ __forceinline__ v8f attn_score_tile(
    const unsigned short* __restrict__ base, const v16u* qa, int kt, int koff,
    int q0, int hf, int lr, bool domask) {
  const unsigned short* kp =
      base + (size_t)(kt * 16 + lr) * (3 * Dc) + koff + hf * 16;
  v16u kb0 = *(const v16u*)(kp);
  v16u kb1 = *(const v16u*)(kp + 32);
  v8f s = (v8f){0, 0, 0, 0, 0, 0, 0, 0};
  s = wmma_bf16(qa[0], kb0, s);
  s = wmma_bf16(qa[1], kb1, s);
  const int key = kt * 16 + lr;
  if (domask) {                                  // diagonal / above tiles only
#pragma unroll
    for (int r = 0; r < 8; r++) {
      const int row = q0 + hf * 8 + r;
      const float sv = s[r] * 0.125f;            // 1/sqrt(64)
      s[r] = (key <= row) ? sv : -10000.0f;
    }
  } else {                                       // strictly-lower tiles
#pragma unroll
    for (int r = 0; r < 8; r++) s[r] *= 0.125f;
  }
  return s;
}

__global__ __launch_bounds__(32) void attn_kernel(
    const unsigned short* __restrict__ qkv, unsigned short* __restrict__ ctx) {
  const int nqt = Tc / 16;
  const int qt  = blockIdx.x % nqt;
  const int hh  = (blockIdx.x / nqt) % Hc;
  const int b   = blockIdx.x / (nqt * Hc);
  const int lane = threadIdx.x;
  const int hf = lane >> 4;
  const int lr = lane & 15;
  const int q0 = qt * 16;
  const int qoff = hh * HDc;
  const int koff = Dc + hh * HDc;
  const int voff = 2 * Dc + hh * HDc;
  const unsigned short* base = qkv + (size_t)b * Tc * (3 * Dc);

  // q fragments (16x64 bf16 -> two 16x32 A-frags), contiguous 32B loads
  v16u qa[2];
  {
    const unsigned short* qp =
        base + (size_t)(q0 + lr) * (3 * Dc) + qoff + hf * 16;
    qa[0] = *(const v16u*)(qp);
    qa[1] = *(const v16u*)(qp + 32);
  }

  // ---- pass 1: row max (causal: kt <= qt only) ----------------------------
  float rmax[8];
#pragma unroll
  for (int r = 0; r < 8; r++) rmax[r] = -3.0e38f;
  for (int kt = 0; kt <= qt; kt++) {
    v8f s = attn_score_tile(base, qa, kt, koff, q0, hf, lr, kt == qt);
#pragma unroll
    for (int r = 0; r < 8; r++) rmax[r] = fmaxf(rmax[r], s[r]);
  }
#pragma unroll
  for (int m = 1; m < 16; m <<= 1)
#pragma unroll
    for (int r = 0; r < 8; r++)
      rmax[r] = fmaxf(rmax[r], __shfl_xor(rmax[r], m, 32));

  // ---- pass 2: exp / row sum / ctx = P @ V (causal chunks only) -----------
  __shared__ __align__(32) unsigned short pb[16 * 32];   // probs, 1 KB
  __shared__ __align__(32) unsigned short Vt[32 * 64];   // V tile, 4 KB
  float rsum[8];
#pragma unroll
  for (int r = 0; r < 8; r++) rsum[r] = 0.f;
  v8f cacc[4];
#pragma unroll
  for (int j = 0; j < 4; j++) cacc[j] = (v8f){0, 0, 0, 0, 0, 0, 0, 0};

  for (int k0 = 0; k0 < q0 + 16; k0 += 32) {
    // stage V tile [32 keys][64 feats] (coalesced 16B copies, key = lane)
    {
      const unsigned short* vp = base + (size_t)(k0 + lane) * (3 * Dc) + voff;
      uint4* d = (uint4*)(Vt + lane * 64);
#pragma unroll
      for (int s2 = 0; s2 < 8; s2++) d[s2] = *(const uint4*)(vp + s2 * 8);
    }
    // probs for 32 keys (second tile of last chunk may be fully masked -> 0)
#pragma unroll
    for (int u = 0; u < 2; u++) {
      const int kt = k0 / 16 + u;
      v8f s = attn_score_tile(base, qa, kt, koff, q0, hf, lr, kt >= qt);
#pragma unroll
      for (int r = 0; r < 8; r++) {
        float p = __expf(s[r] - rmax[r]);
        rsum[r] += p;
        pb[(hf * 8 + r) * 32 + u * 16 + lr] = f2bf(p);   // [q row][key]
      }
    }
    __syncthreads();
    v16u pa = *(const v16u*)(pb + lr * 32 + hf * 16);    // A-frag of P(16x32)
    v16u vb[4];                                          // B-frags of V(32x16)
#pragma unroll
    for (int j = 0; j < 4; j++)
#pragma unroll
      for (int e = 0; e < 16; e++)
        vb[j][e] = Vt[(hf * 16 + e) * 64 + j * 16 + lr];
    __syncthreads();
#pragma unroll
    for (int j = 0; j < 4; j++) cacc[j] = wmma_bf16(pa, vb[j], cacc[j]);
  }

#pragma unroll
  for (int m = 1; m < 16; m <<= 1)
#pragma unroll
    for (int r = 0; r < 8; r++) rsum[r] += __shfl_xor(rsum[r], m, 32);

  // ---- store ctx [B,T,D] as bf16 (feeds proj GEMM) ------------------------
#pragma unroll
  for (int j = 0; j < 4; j++)
#pragma unroll
    for (int r = 0; r < 8; r++) {
      const int row = q0 + hf * 8 + r;
      ctx[((size_t)b * Tc + row) * Dc + hh * HDc + j * 16 + lr] =
          f2bf(cacc[j][r] / rsum[r]);
    }
}

// ---------------------------------------------------------------------------
// Host-side orchestration (graph-capture safe).
// ---------------------------------------------------------------------------
extern "C" void kernel_launch(void* const* d_in, const int* in_sizes, int n_in,
                              void* d_out, int out_size, void* d_ws,
                              size_t ws_size, hipStream_t stream) {
  (void)in_sizes; (void)n_in; (void)out_size; (void)ws_size;

  const float* inputs_embeds = (const float*)d_in[0];
  const float* ln1_w  = (const float*)d_in[1];
  const float* ln1_b  = (const float*)d_in[2];
  const float* W_qkv  = (const float*)d_in[3];
  const float* b_qkv  = (const float*)d_in[4];
  const float* W_proj = (const float*)d_in[5];
  const float* b_proj = (const float*)d_in[6];
  const float* ln2_w  = (const float*)d_in[7];
  const float* ln2_b  = (const float*)d_in[8];
  const float* W_fc   = (const float*)d_in[9];
  const float* b_fc   = (const float*)d_in[10];
  const float* W_fc2  = (const float*)d_in[11];
  const float* b_fc2  = (const float*)d_in[12];
  const float* lnf_w  = (const float*)d_in[13];
  const float* lnf_b  = (const float*)d_in[14];

  // ---- workspace layout ----------------------------------------------------
  char* ws = (char*)d_ws;
  size_t off = 0;
  auto take = [&](size_t bytes) {
    char* p = ws + off;
    off += (bytes + 255) & ~(size_t)255;
    return p;
  };
  float*          x    = (float*)         take((size_t)MTc * Dc * 4);       // residual (fp32)
  unsigned short* xn   = (unsigned short*)take((size_t)MTc * Dc * 2);       // LN out (bf16)
  unsigned short* qkv  = (unsigned short*)take((size_t)MTc * 3 * Dc * 2);   // bf16
  unsigned short* ctx  = (unsigned short*)take((size_t)MTc * Dc * 2);       // bf16
  unsigned short* hb   = (unsigned short*)take((size_t)MTc * DFc * 2);      // bf16
  unsigned short* Wq   = (unsigned short*)take((size_t)Lc * Dc * 3 * Dc * 2);
  unsigned short* Wp   = (unsigned short*)take((size_t)Lc * Dc * Dc * 2);
  unsigned short* Wf   = (unsigned short*)take((size_t)Lc * Dc * DFc * 2);
  unsigned short* Wf2  = (unsigned short*)take((size_t)Lc * DFc * Dc * 2);

  const int MT = MTc;
  const dim3 blk256(256), blk32(32);

  // ---- one-time per-launch weight conversion to bf16 ----------------------
  cvt_kernel<<<4096, blk256, 0, stream>>>(W_qkv,  Wq,  (long)Lc * Dc * 3 * Dc / 4);
  cvt_kernel<<<4096, blk256, 0, stream>>>(W_proj, Wp,  (long)Lc * Dc * Dc / 4);
  cvt_kernel<<<4096, blk256, 0, stream>>>(W_fc,   Wf,  (long)Lc * Dc * DFc / 4);
  cvt_kernel<<<4096, blk256, 0, stream>>>(W_fc2,  Wf2, (long)Lc * DFc * Dc / 4);

  hipMemcpyAsync(x, inputs_embeds, (size_t)MT * Dc * sizeof(float),
                 hipMemcpyDeviceToDevice, stream);

  for (int l = 0; l < Lc; l++) {
    // ---- attention block ----
    ln_kernel<<<MT, blk256, 0, stream>>>(x, ln1_w + (size_t)l * Dc,
                                         ln1_b + (size_t)l * Dc, nullptr, xn, Dc);
    gemm_bf16_kernel<1><<<dim3(3 * Dc / BN, MT / BM), blk256, 0, stream>>>(
        xn, Wq + (size_t)l * Dc * 3 * Dc, b_qkv + (size_t)l * 3 * Dc,
        nullptr, nullptr, qkv, MT, 3 * Dc, Dc);
    attn_kernel<<<Bc * Hc * (Tc / 16), blk32, 0, stream>>>(qkv, ctx);
    gemm_bf16_kernel<0><<<dim3(Dc / BN, MT / BM), blk256, 0, stream>>>(
        ctx, Wp + (size_t)l * Dc * Dc, b_proj + (size_t)l * Dc,
        /*Res=*/x, /*OutF=*/x, nullptr, MT, Dc, Dc);

    // ---- MLP block ----
    ln_kernel<<<MT, blk256, 0, stream>>>(x, ln2_w + (size_t)l * Dc,
                                         ln2_b + (size_t)l * Dc, nullptr, xn, Dc);
    gemm_bf16_kernel<2><<<dim3(DFc / BN, MT / BM), blk256, 0, stream>>>(
        xn, Wf + (size_t)l * Dc * DFc, b_fc + (size_t)l * DFc,
        nullptr, nullptr, hb, MT, DFc, Dc);
    gemm_bf16_kernel<0><<<dim3(Dc / BN, MT / BM), blk256, 0, stream>>>(
        hb, Wf2 + (size_t)l * DFc * Dc, b_fc2 + (size_t)l * Dc,
        /*Res=*/x, /*OutF=*/x, nullptr, MT, Dc, DFc);
  }

  ln_kernel<<<MT, blk256, 0, stream>>>(x, lnf_w, lnf_b, (float*)d_out, nullptr, Dc);
}
